// StructuredTransformerEncoderBlock_10041633538561
// MI455X (gfx1250) — compile-verified, matrix operand order checked by reading, not run on hardware
//
#include <hip/hip_runtime.h>

// ---------------------------------------------------------------------------
// Problem constants (match reference)
// ---------------------------------------------------------------------------
#define SIZE   256
#define DIST   64
#define KNBR   16
#define HEADS  8
#define ATTN   32
#define HIDDEN 512
#define HA     256
#define EDGE   320   // SIZE + DIST

typedef __attribute__((ext_vector_type(16))) __bf16 v16bf;
typedef __attribute__((ext_vector_type(8)))  float  v8f;

// ---------------------------------------------------------------------------
// WMMA helpers (CDNA5 bf16 16x16x32, f32 accumulate)
// ---------------------------------------------------------------------------
static __device__ __forceinline__ v8f wmma_bf16(v16bf a, v16bf b, v8f c) {
    return __builtin_amdgcn_wmma_f32_16x16x32_bf16(
        /*neg_a=*/false, a, /*neg_b=*/false, b,
        /*c_mod=*/(short)0, c, /*reuse_a=*/false, /*reuse_b=*/false);
}

// Load a 16-element bf16 fragment for one lane. `p` points at the start of the
// 32-wide K-slice for this lane's row (A) / column (B); `half` = lane>>4.
// ISA 16-bit A layout: lanes 0-15 hold K {0..7,16..23}, lanes 16-31 hold
// K {8..15,24..31}; B mirrors this per column.
static __device__ __forceinline__ v16bf frag_from(const __bf16* p, int half) {
    v16bf f;
#pragma unroll
    for (int i = 0; i < 8; ++i) f[i] = p[half * 8 + i];
#pragma unroll
    for (int i = 0; i < 8; ++i) f[8 + i] = p[16 + half * 8 + i];
    return f;
}

// ---------------------------------------------------------------------------
// Weight transpose + f32->bf16 cast:  Wt[o][i] = (bf16) W[i][o]
// ---------------------------------------------------------------------------
__global__ void k_cast_transpose(const float* __restrict__ W,
                                 __bf16* __restrict__ Wt,
                                 int rows_in, int cols_out) {
    int i = blockIdx.x * blockDim.x + threadIdx.x;
    int total = rows_in * cols_out;
    if (i < total) {
        int o = i % cols_out;
        int r = i / cols_out;
        Wt[(size_t)o * rows_in + r] = (__bf16)W[i];
    }
}

// features -> bf16 (raw, for Q) and bf16(relu) (for MLP)
__global__ void k_cast_features(const float* __restrict__ f,
                                __bf16* __restrict__ Xb,
                                __bf16* __restrict__ Xr, int total) {
    int i = blockIdx.x * blockDim.x + threadIdx.x;
    if (i < total) {
        float x = f[i];
        Xb[i] = (__bf16)x;
        Xr[i] = (__bf16)fmaxf(x, 0.0f);
    }
}

// ---------------------------------------------------------------------------
// Generic WMMA GEMM:  out[M,Nd] = epilogue(A[M,Kd] @ Wt^T + bias)
//   A   : bf16 row-major [M,Kd]
//   Wt  : bf16 transposed weights [Nd,Kd] (row per output column)
//   64x64 block tile, 128 threads = 4 wave32; wave w -> rows [w*16, w*16+16)
//   ACT: relu if 1.  OBF: bf16 output if 1 else f32.  RES: add resid (stride Nd)
// ---------------------------------------------------------------------------
template <int ACT, int OBF, int RES>
__global__ __launch_bounds__(128) void k_gemm_wmma(
    const __bf16* __restrict__ A, const __bf16* __restrict__ Wt,
    const float* __restrict__ bias, void* __restrict__ out,
    const float* __restrict__ resid, int M, int Kd, int Nd) {
    __shared__ __bf16 sA[64 * 32];

    const int tid  = threadIdx.x;
    const int lane = tid & 31;
    const int wave = tid >> 5;
    const int m0   = blockIdx.x * 64;
    const int n0   = blockIdx.y * 64;
    const int half = lane >> 4;
    const int lcol = lane & 15;

    v8f acc[4];
#pragma unroll
    for (int j = 0; j < 4; ++j)
#pragma unroll
        for (int r = 0; r < 8; ++r) acc[j][r] = 0.0f;

    for (int k0 = 0; k0 < Kd; k0 += 32) {
        __syncthreads();
        {   // stage 64x32 A tile: each thread copies 16 contiguous bf16 (32 B)
            int r = tid >> 1;
            int c = (tid & 1) * 16;
            const __bf16* src = A + (size_t)(m0 + r) * Kd + k0 + c;
            *(uint4*)(&sA[r * 32 + c])     = *(const uint4*)(src);
            *(uint4*)(&sA[r * 32 + c + 8]) = *(const uint4*)(src + 8);
        }
        __syncthreads();

        v16bf af = frag_from(&sA[(wave * 16 + lcol) * 32], half);
#pragma unroll
        for (int j = 0; j < 4; ++j) {
            const __bf16* bp =
                Wt + (size_t)(n0 + j * 16 + lcol) * Kd + k0;
            v16bf bf = frag_from(bp, half);
            acc[j] = wmma_bf16(af, bf, acc[j]);
        }
    }

    // Epilogue: C layout = VGPR r -> M = (lane<16 ? r : r+8), N = lane&15
#pragma unroll
    for (int j = 0; j < 4; ++j) {
        int col = n0 + j * 16 + lcol;
        float bia = bias[col];
#pragma unroll
        for (int r = 0; r < 8; ++r) {
            int row = m0 + wave * 16 + ((lane < 16) ? r : r + 8);
            float x = acc[j][r] + bia;
            if (ACT) x = fmaxf(x, 0.0f);
            if (RES) x += resid[(size_t)row * Nd + col];
            if (OBF) ((__bf16*)out)[(size_t)row * Nd + col] = (__bf16)x;
            else     ((float*)out)[(size_t)row * Nd + col] = x;
        }
    }
}

// ---------------------------------------------------------------------------
// Fused edge kernel: one block (128 thr = 4 wave32) per node.
//  1) gather 16 neighbour rows of `local` + bf16(distances) -> 16x320 LDS tile
//  2) K = edge @ Wk, V = edge @ Wv via WMMA (10 k-steps of 32), kept in LDS
//  3) per-head scores vs q, softmax over 16 neighbours, weighted sum of V
//  4) write o[n, 256] as bf16
// ---------------------------------------------------------------------------
__global__ __launch_bounds__(128) void k_edge_attn(
    const __bf16* __restrict__ localb, const float* __restrict__ dist,
    const int* __restrict__ structure, const float* __restrict__ Qf,
    const __bf16* __restrict__ Wkt, const __bf16* __restrict__ Wvt,
    const float* __restrict__ bk, const float* __restrict__ bv,
    __bf16* __restrict__ Ob) {
    __shared__ __bf16 sE[16 * EDGE];       // 10 KB edge tile
    __shared__ float  sKV[2][16 * HA];     // 32 KB  K then V (f32)
    __shared__ float  sQ[HA];
    __shared__ float  sAttn[HEADS][KNBR];

    const int n    = blockIdx.x;
    const int tid  = threadIdx.x;
    const int lane = tid & 31;
    const int wave = tid >> 5;
    const int half = lane >> 4;
    const int lcol = lane & 15;

    // q row for this node
    sQ[tid]       = Qf[(size_t)n * HA + tid];
    sQ[tid + 128] = Qf[(size_t)n * HA + tid + 128];

    {   // build edge tile: 8 threads per edge row
        int e   = tid >> 3;   // 0..15
        int sub = tid & 7;    // 0..7
        int idx = structure[n * KNBR + e];
        const __bf16* lsrc = localb + (size_t)idx * SIZE + sub * 32;
        __bf16* ldst = &sE[e * EDGE + sub * 32];
#pragma unroll
        for (int i = 0; i < 4; ++i)
            ((uint4*)ldst)[i] = ((const uint4*)lsrc)[i];
        const float* dsrc = dist + ((size_t)n * KNBR + e) * DIST + sub * 8;
        __bf16* ddst = &sE[e * EDGE + SIZE + sub * 8];
#pragma unroll
        for (int i = 0; i < 8; ++i) ddst[i] = (__bf16)dsrc[i];
    }
    __syncthreads();

    // --- K/V GEMM: 32 (mat, subtile) pairs over 4 waves, 8 each ------------
    v8f acc[8];
#pragma unroll
    for (int j = 0; j < 8; ++j)
#pragma unroll
        for (int r = 0; r < 8; ++r) acc[j][r] = 0.0f;

    for (int kc = 0; kc < EDGE / 32; ++kc) {
        v16bf af = frag_from(&sE[lcol * EDGE + kc * 32], half);
#pragma unroll
        for (int jj = 0; jj < 8; ++jj) {
            int p   = wave * 8 + jj;
            int om  = p >> 4;           // 0 = K, 1 = V
            int sub = p & 15;           // column subtile
            const __bf16* wt = om ? Wvt : Wkt;
            v16bf bf = frag_from(
                wt + (size_t)(sub * 16 + lcol) * EDGE + kc * 32, half);
            acc[jj] = wmma_bf16(af, bf, acc[jj]);
        }
    }

#pragma unroll
    for (int jj = 0; jj < 8; ++jj) {
        int p   = wave * 8 + jj;
        int om  = p >> 4;
        int sub = p & 15;
        int col = sub * 16 + lcol;
        float bia = om ? bv[col] : bk[col];
#pragma unroll
        for (int r = 0; r < 8; ++r) {
            int m = (lane < 16) ? r : r + 8;   // edge index
            sKV[om][m * HA + col] = acc[jj][r] + bia;
        }
    }
    __syncthreads();

    // --- attention: wave w -> heads 2w (lanes 0-15) and 2w+1 (lanes 16-31) -
    {
        int h = wave * 2 + half;
        int e = lcol;
        float s = 0.0f;
#pragma unroll
        for (int a = 0; a < ATTN; ++a)
            s += sQ[h * ATTN + a] * sKV[0][e * HA + h * ATTN + a];
        s *= 0.17677669529663688f;   // 1/sqrt(32)

        float mx = s;
#pragma unroll
        for (int off = 8; off; off >>= 1)
            mx = fmaxf(mx, __shfl_xor(mx, off, 16));
        float ex = __expf(s - mx);
        float sm = ex;
#pragma unroll
        for (int off = 8; off; off >>= 1)
            sm += __shfl_xor(sm, off, 16);
        sAttn[h][e] = ex / sm;
    }
    __syncthreads();

    // --- weighted sum of V; lane = attention dim, 2 heads per wave ---------
#pragma unroll
    for (int hh = 0; hh < 2; ++hh) {
        int hd = wave * 2 + hh;
        float o = 0.0f;
#pragma unroll
        for (int e = 0; e < KNBR; ++e)
            o += sAttn[hd][e] * sKV[1][e * HA + hd * ATTN + lane];
        Ob[(size_t)n * HA + hd * ATTN + lane] = (__bf16)o;
    }
}

// ---------------------------------------------------------------------------
// Host launcher
// ---------------------------------------------------------------------------
extern "C" void kernel_launch(void* const* d_in, const int* in_sizes, int n_in,
                              void* d_out, int out_size, void* d_ws,
                              size_t ws_size, hipStream_t stream) {
    const float* features  = (const float*)d_in[0];
    const float* distances = (const float*)d_in[1];
    const int*   structure = (const int*)d_in[2];
    const float* W0 = (const float*)d_in[3];
    const float* b0 = (const float*)d_in[4];
    const float* W1 = (const float*)d_in[5];
    const float* b1 = (const float*)d_in[6];
    const float* W2 = (const float*)d_in[7];
    const float* b2 = (const float*)d_in[8];
    const float* Wq = (const float*)d_in[9];
    const float* bq = (const float*)d_in[10];
    const float* Wk = (const float*)d_in[11];
    const float* bk = (const float*)d_in[12];
    const float* Wv = (const float*)d_in[13];
    const float* bv = (const float*)d_in[14];
    const float* Wo = (const float*)d_in[15];
    const float* bo = (const float*)d_in[16];
    float* out = (float*)d_out;

    const int N = in_sizes[0] / SIZE;

    // ---- workspace carve-out ----
    char* ws = (char*)d_ws;
    size_t off = 0;
    auto carve = [&](size_t bytes) -> char* {
        char* p = ws + off;
        off = (off + bytes + 255) & ~(size_t)255;
        return p;
    };
    __bf16* W0t = (__bf16*)carve((size_t)SIZE * HIDDEN * 2);
    __bf16* W1t = (__bf16*)carve((size_t)HIDDEN * HIDDEN * 2);
    __bf16* W2t = (__bf16*)carve((size_t)HIDDEN * SIZE * 2);
    __bf16* Wqt = (__bf16*)carve((size_t)SIZE * HA * 2);
    __bf16* Wkt = (__bf16*)carve((size_t)EDGE * HA * 2);
    __bf16* Wvt = (__bf16*)carve((size_t)EDGE * HA * 2);
    __bf16* Wot = (__bf16*)carve((size_t)HA * SIZE * 2);
    __bf16* Xb  = (__bf16*)carve((size_t)N * SIZE * 2);
    __bf16* Xr  = (__bf16*)carve((size_t)N * SIZE * 2);
    __bf16* H1  = (__bf16*)carve((size_t)N * HIDDEN * 2);
    __bf16* H2  = (__bf16*)carve((size_t)N * HIDDEN * 2);
    __bf16* LOC = (__bf16*)carve((size_t)N * SIZE * 2);
    float*  Qf  = (float*) carve((size_t)N * HA * 4);
    __bf16* Ob  = (__bf16*)carve((size_t)N * HA * 2);

    auto cdiv = [](int a, int b) { return (a + b - 1) / b; };

    // ---- weight transpose/cast ----
    k_cast_transpose<<<cdiv(SIZE * HIDDEN, 256), 256, 0, stream>>>(W0, W0t, SIZE, HIDDEN);
    k_cast_transpose<<<cdiv(HIDDEN * HIDDEN, 256), 256, 0, stream>>>(W1, W1t, HIDDEN, HIDDEN);
    k_cast_transpose<<<cdiv(HIDDEN * SIZE, 256), 256, 0, stream>>>(W2, W2t, HIDDEN, SIZE);
    k_cast_transpose<<<cdiv(SIZE * HA, 256), 256, 0, stream>>>(Wq, Wqt, SIZE, HA);
    k_cast_transpose<<<cdiv(EDGE * HA, 256), 256, 0, stream>>>(Wk, Wkt, EDGE, HA);
    k_cast_transpose<<<cdiv(EDGE * HA, 256), 256, 0, stream>>>(Wv, Wvt, EDGE, HA);
    k_cast_transpose<<<cdiv(HA * SIZE, 256), 256, 0, stream>>>(Wo, Wot, HA, SIZE);

    // ---- feature cast (raw + relu) ----
    k_cast_features<<<cdiv(N * SIZE, 256), 256, 0, stream>>>(features, Xb, Xr, N * SIZE);

    // ---- local MLP (relu between + after) ----
    k_gemm_wmma<1, 1, 0><<<dim3(N / 64, HIDDEN / 64), 128, 0, stream>>>(
        Xr, W0t, b0, H1, nullptr, N, SIZE, HIDDEN);
    k_gemm_wmma<1, 1, 0><<<dim3(N / 64, HIDDEN / 64), 128, 0, stream>>>(
        H1, W1t, b1, H2, nullptr, N, HIDDEN, HIDDEN);
    k_gemm_wmma<1, 1, 0><<<dim3(N / 64, SIZE / 64), 128, 0, stream>>>(
        H2, W2t, b2, LOC, nullptr, N, HIDDEN, SIZE);

    // ---- Q projection (raw features) ----
    k_gemm_wmma<0, 0, 0><<<dim3(N / 64, HA / 64), 128, 0, stream>>>(
        Xb, Wqt, bq, Qf, nullptr, N, SIZE, HA);

    // ---- fused gather + K/V WMMA + softmax attention ----
    k_edge_attn<<<N, 128, 0, stream>>>(LOC, distances, structure, Qf,
                                       Wkt, Wvt, bk, bv, Ob);

    // ---- output projection + residual ----
    k_gemm_wmma<0, 0, 1><<<dim3(N / 64, SIZE / 64), 128, 0, stream>>>(
        Ob, Wot, bo, out, features, N, HA, SIZE);
}
